// KPConv_26225070309983
// MI455X (gfx1250) — matrix operand order repaired and checked by reading, not compiled, not run on hardware
//
#include <hip/hip_runtime.h>

// ---------------------------------------------------------------------------
// KPConv fused kernel for gfx1250 (MI455X), wave32 + WMMA bf16.
//  - s_feats pre-converted to bf16 in d_ws: halves gathered HBM traffic
//    (491 MB -> 245 MB), which is the roofline-dominant term.
//  - second-GEMM contraction flattened as j = c*16 + k so phase-B D tiles
//    store to LDS as single b128 writes.
//  - LDS rows padded for conflict-free WMMA fragment loads.
// ---------------------------------------------------------------------------

typedef __attribute__((ext_vector_type(16))) __bf16 v16bf;
typedef __attribute__((ext_vector_type(8)))  __bf16 v8bf;
typedef __attribute__((ext_vector_type(8)))  float  v8f;

#define MPTS   30000
#define NPTS   30000
#define HNB    32
#define HPAD   40            // +8 pad: 80B row stride -> conflict-free fragment loads
#define KPTS   15
#define KPAD   16
#define CIN    128
#define COUT   128
#define SIGMA  0.9f
#define QT     16            // queries per block
#define NBLK   (MPTS / QT)   // 1875, exact

#define JDIM   2048          // flattened contraction: j = c*16 + k  (c:128, k:16)
#define JPAD   2056          // +8 pad: 4112B row stride -> conflict-free A loads
#define NJC    (JDIM / 32)   // 64 contraction chunks of 32

// Packed weights: 8 d-tiles * 64 j-chunks * (32 lanes * 16 elems)
#define WB_ELEMS (8 * NJC * 512)
#define WB_BYTES (WB_ELEMS * 2)          // 512 KB, 16B-aligned carve of d_ws
#define SF_ELEMS (NPTS * CIN)            // bf16 feature copy: 7.68 MB

// ---------------------------------------------------------------------------
// Pack weights [15][128][128] f32 -> bf16 WMMA-B tiles over contraction j.
// Tile (dt, jc): B rows are j = jc*32 .. +31 (j = c*16 + k), cols d = dt*16..+15.
// Lane layout (16-bit B 32x16): col N = lane&15, elem e: row K =
// (e&7) + (lane>>4)*8 + (e>>3)*16.  Packed so each lane's 16 bf16 are contiguous (32B).
// Rows with k==15 are zero (pad kernel point).
// ---------------------------------------------------------------------------
__global__ void kpconv_pack_weights(const float* __restrict__ w,
                                    __bf16* __restrict__ wb) {
    int idx = blockIdx.x * blockDim.x + threadIdx.x;
    if (idx >= WB_ELEMS) return;
    int e    = idx & 15;
    int lane = (idx >> 4) & 31;
    int tile = idx >> 9;
    int jc   = tile & (NJC - 1);  // contraction chunk
    int dt   = tile >> 6;         // d-tile (16 outputs each)
    int hi   = lane >> 4;
    int dd   = lane & 15;
    int j    = jc * 32 + (e & 7) + hi * 8 + (e >> 3) * 16;
    int c    = j >> 4;            // input channel
    int k    = j & 15;            // kernel point (15 == pad)
    float v  = (k < KPTS) ? w[(k * CIN + c) * COUT + dt * 16 + dd] : 0.0f;
    wb[idx] = (__bf16)v;
}

// ---------------------------------------------------------------------------
// One-time f32 -> bf16 conversion of the support features (streamed once;
// each row is re-gathered ~32x afterwards, so this halves total HBM bytes).
// ---------------------------------------------------------------------------
__global__ void kpconv_cvt_feats(const float* __restrict__ f,
                                 __bf16* __restrict__ fb) {
    long idx = (long)(blockIdx.x * blockDim.x + threadIdx.x) * 8;
    if (idx >= (long)SF_ELEMS) return;
    float4 a = *(const float4*)&f[idx];
    float4 b = *(const float4*)&f[idx + 4];
    v8bf o;
    o[0] = (__bf16)a.x; o[1] = (__bf16)a.y; o[2] = (__bf16)a.z; o[3] = (__bf16)a.w;
    o[4] = (__bf16)b.x; o[5] = (__bf16)b.y; o[6] = (__bf16)b.z; o[7] = (__bf16)b.w;
    *(v8bf*)&fb[idx] = o;
}

// ---------------------------------------------------------------------------
// Fused KPConv: influence weights -> gather -> wf = w @ f -> out = wf @ W
// ---------------------------------------------------------------------------
__launch_bounds__(256)
__global__ void kpconv_fused(const float* __restrict__ q_pts,
                             const float* __restrict__ s_pts,
                             const __bf16* __restrict__ sfb,
                             const int*   __restrict__ inds,
                             const float* __restrict__ kpts,
                             const __bf16* __restrict__ wb,
                             float* __restrict__ out) {
    __shared__ __bf16 wLds [QT][KPAD][HPAD];  // 20 KB influence weights
    __shared__ __bf16 fLds [QT][32][HPAD];    // 40 KB gathered feats (c-major)
    __shared__ __bf16 wfLds[QT][JPAD];        // ~64 KB weighted feats, j-flattened
    __shared__ int    nLds [QT][HNB];         //  2 KB neighbor indices

    const int t     = threadIdx.x;
    const int lane  = t & 31;
    const int wave  = t >> 5;        // 8 waves
    const int qbase = blockIdx.x * QT;
    const int hi    = lane >> 4;     // half-wave select
    const int row   = lane & 15;

    // ---------------- Phase 0: influence weights w[q][k][h] ----------------
    for (int step = 0; step < 2; ++step) {
        int pair = t + step * 256;
        int q = pair >> 5;
        int h = pair & 31;
        int qg = qbase + q;
        int n = inds[qg * HNB + h];
        nLds[q][h] = n;
        float qx = q_pts[qg * 3 + 0];
        float qy = q_pts[qg * 3 + 1];
        float qz = q_pts[qg * 3 + 2];
        bool valid = (n >= 0) && (n < NPTS);
        float sx = valid ? s_pts[n * 3 + 0] : 1.0e6f;
        float sy = valid ? s_pts[n * 3 + 1] : 1.0e6f;
        float sz = valid ? s_pts[n * 3 + 2] : 1.0e6f;
        float rx = sx - qx, ry = sy - qy, rz = sz - qz;
        #pragma unroll
        for (int k = 0; k < KPTS; ++k) {
            float dx = rx - kpts[k * 3 + 0];
            float dy = ry - kpts[k * 3 + 1];
            float dz = rz - kpts[k * 3 + 2];
            float d2 = dx * dx + dy * dy + dz * dz;
            float wk = fmaxf(1.0f - sqrtf(d2) * (1.0f / SIGMA), 0.0f);
            wLds[q][k][h] = (__bf16)wk;
        }
        wLds[q][KPTS][h] = (__bf16)0.0f;   // zero pad row k=15
    }
    __syncthreads();

    // --------- Phases A+B per 32-channel chunk: gather + wf WMMAs ----------
    for (int cc = 0; cc < 4; ++cc) {
        // Gather: 4 threads per (q,h) pair cover 32 bf16 channels (b128 each);
        // rows are random but each per-row segment is a 64B contiguous read.
        {
            int sub   = t & 3;
            int pbase = t >> 2;          // 0..63
            for (int it = 0; it < 8; ++it) {
                int pair = it * 64 + pbase;
                int q = pair >> 5;
                int h = pair & 31;
                int n = nLds[q][h];
                v8bf v = {};
                if (n >= 0 && n < NPTS) {
                    const __bf16* src = &sfb[n * CIN + cc * 32 + sub * 8];
                    v = *(const v8bf*)src;
                    if (cc < 3) __builtin_prefetch(src + 32, 0, 0); // next chunk
                }
                int c0 = sub * 8;
                #pragma unroll
                for (int j = 0; j < 8; ++j) fLds[q][c0 + j][h] = v[j];
            }
        }
        __syncthreads();

        // Phase B: wf[q][:, c-chunk] = w[q] (16k x 32h) @ fT[q] (32h x 32c)
        #pragma unroll
        for (int qi = 0; qi < 2; ++qi) {
            int q = wave * 2 + qi;
            union { v16bf v; v8bf p[2]; } A;
            // A 16x32: lane row k=lane&15; e<8 -> h=e+hi*8 ; e>=8 -> h=e+8+hi*8
            A.p[0] = *(const v8bf*)&wLds[q][row][hi * 8];
            A.p[1] = *(const v8bf*)&wLds[q][row][16 + hi * 8];
            #pragma unroll
            for (int ct = 0; ct < 2; ++ct) {
                union { v16bf v; v8bf p[2]; } B;
                int c = cc * 32 + ct * 16 + row;    // global channel (B col)
                // B 32x16: e<8 -> h=e+hi*8 ; e>=8 -> h=e+8+hi*8
                B.p[0] = *(const v8bf*)&fLds[q][ct * 16 + row][hi * 8];
                B.p[1] = *(const v8bf*)&fLds[q][ct * 16 + row][16 + hi * 8];
                v8f d = {};
                d = __builtin_amdgcn_wmma_f32_16x16x32_bf16(
                        false, A.v, false, B.v, (short)0, d, false, false);
                // D: lane col c = lane&15; vgpr p -> row k = p + hi*8.
                // j = c*16 + k -> lane's 8 D values contiguous: one b128 store.
                v8bf pk;
                #pragma unroll
                for (int p = 0; p < 8; ++p) pk[p] = (__bf16)d[p];
                *(v8bf*)&wfLds[q][c * 16 + hi * 8] = pk;
            }
        }
        __syncthreads();
    }

    // -------- Phase C: out[q][d] = sum_j wf[q][j] * Wb[j][d] ---------------
    // wave owns d-tile = wave (16 outputs); 64 contraction chunks of 32.
    v8f acc = {};
    for (int jc = 0; jc < NJC; ++jc) {
        union { v16bf v; v8bf p[2]; } A;
        union { v16bf v; v8bf p[2]; } B;
        // A 16x32: lane row q=lane&15; elem e -> j_local = (e<8?e:e+8)+hi*8
        A.p[0] = *(const v8bf*)&wfLds[row][jc * 32 + hi * 8];
        A.p[1] = *(const v8bf*)&wfLds[row][jc * 32 + 16 + hi * 8];
        // B tiles pre-packed: each lane's 16 bf16 contiguous (32 B), L2-hot
        const __bf16* bp = wb + (((wave * NJC + jc) << 9) + lane * 16);
        B.p[0] = *(const v8bf*)bp;
        B.p[1] = *(const v8bf*)(bp + 8);
        acc = __builtin_amdgcn_wmma_f32_16x16x32_bf16(
                  false, A.v, false, B.v, (short)0, acc, false, false);
    }
    // D layout: lane col d = wave*16 + (lane&15); vgpr p -> row q = p + hi*8
    #pragma unroll
    for (int p = 0; p < 8; ++p) {
        int q = p + hi * 8;
        out[(qbase + q) * COUT + wave * 16 + row] = acc[p];
    }
}

// ---------------------------------------------------------------------------
extern "C" void kernel_launch(void* const* d_in, const int* in_sizes, int n_in,
                              void* d_out, int out_size, void* d_ws, size_t ws_size,
                              hipStream_t stream) {
    const float* q_pts   = (const float*)d_in[0];
    const float* s_pts   = (const float*)d_in[1];
    const float* s_feats = (const float*)d_in[2];
    const int*   inds    = (const int*)  d_in[3];
    const float* kpts    = (const float*)d_in[4];
    const float* weights = (const float*)d_in[5];
    float*   out = (float*)d_out;

    __bf16* wb  = (__bf16*)d_ws;                       // 512 KB packed weights
    __bf16* sfb = (__bf16*)((char*)d_ws + WB_BYTES);   // 7.68 MB bf16 features

    kpconv_pack_weights<<<(WB_ELEMS + 255) / 256, 256, 0, stream>>>(weights, wb);
    kpconv_cvt_feats<<<(SF_ELEMS / 8 + 255) / 256, 256, 0, stream>>>(s_feats, sfb);
    kpconv_fused<<<NBLK, 256, 0, stream>>>(q_pts, s_pts, sfb, inds, kpts, wb, out);
}
